// StandardSchNet_31559419691086
// MI455X (gfx1250) — matrix operand order repaired and checked by reading, not compiled, not run on hardware
//
#include <hip/hip_runtime.h>
#include <math.h>

// SchNet interaction block for MI455X (gfx1250, wave32).
// - All GEMMs via v_wmma_f32_16x16x32_f16 (f16 in, f32 accumulate).
// - Weight (B) fragments register-resident per wave (loaded once, L2-hot).
// - edge_attr tiles DMA'd into LDS with the Tensor Data Mover (double-buffered,
//   s_wait_tensorcnt pipelined), since edge_attr is the only real HBM stream.

typedef __attribute__((ext_vector_type(16))) _Float16 v16h;
typedef __attribute__((ext_vector_type(4)))  _Float16 h4;
typedef __attribute__((ext_vector_type(8)))  float    v8f;
typedef __attribute__((ext_vector_type(4)))  float    f4;
typedef __attribute__((ext_vector_type(4)))  unsigned int u32x4;
typedef __attribute__((ext_vector_type(8)))  unsigned int u32x8;

#define FDIM 128
#define RBFD 64
#define CUT  5.0f

// ---------------- WMMA fragment helpers (CDNA5 ISA 7.12.2, wave32) ----------------

// A fragment: 16x32 (MxK) from f16 LDS tile (row-major, given stride), k offset k0.
__device__ __forceinline__ v16h load_a_frag_h(const _Float16* s, int stride, int k0, int lane) {
  const int g = lane >> 4, m = lane & 15;
  v16h a;
#pragma unroll
  for (int e = 0; e < 16; ++e) {
    const int k = ((e < 8) ? e : (e + 8)) + 8 * g + k0;
    a[e] = s[m * stride + k];
  }
  return a;
}

// A fragment from an f32 LDS tile (TDM copies raw f32), converting to f16.
__device__ __forceinline__ v16h load_a_frag_f(const float* s, int stride, int k0, int lane) {
  const int g = lane >> 4, m = lane & 15;
  v16h a;
#pragma unroll
  for (int e = 0; e < 16; ++e) {
    const int k = ((e < 8) ? e : (e + 8)) + 8 * g + k0;
    a[e] = (_Float16)s[m * stride + k];
  }
  return a;
}

// B fragment: 32x16 (KxN) slice of a row-major f32 weight (ld = row stride),
// loaded ONCE per wave straight from global (weights are L2-resident).
__device__ __forceinline__ v16h load_b_frag_g(const float* __restrict__ w, int ld,
                                              int k0, int n0, int lane) {
  const int g = lane >> 4, n = lane & 15;
  v16h b;
#pragma unroll
  for (int e = 0; e < 16; ++e) {
    const int k = ((e < 8) ? e : (e + 8)) + 8 * g + k0;
    b[e] = (_Float16)w[k * ld + n0 + n];
  }
  return b;
}

__device__ __forceinline__ v8f wmma_f16(v16h a, v16h b, v8f c) {
  return __builtin_amdgcn_wmma_f32_16x16x32_f16(false, a, false, b, (short)0, c, false, false);
}

__device__ __forceinline__ h4 cvt4(f4 v) {
  h4 r;
  r[0] = (_Float16)v[0]; r[1] = (_Float16)v[1];
  r[2] = (_Float16)v[2]; r[3] = (_Float16)v[3];
  return r;
}

// ---------------- Tensor Data Mover: 2D tile load (D# per ISA 08_async_tensor) ----
// tensor: rows_total x dim0 (f32, row stride = dim0); tile: tile1 rows x tile0 elems.
__device__ __forceinline__ void tdm_load_2d(unsigned lds_off, const float* gsrc,
                                            unsigned dim0, unsigned rows_total,
                                            unsigned tile0, unsigned tile1) {
  unsigned long long ga = (unsigned long long)(const void*)gsrc;
  u32x4 g0;
  g0[0] = 1u;                                            // count=1, user descriptor
  g0[1] = lds_off;                                       // lds_addr (bytes)
  g0[2] = (unsigned)(ga & 0xFFFFFFFFu);                  // global_addr[31:0]
  g0[3] = (unsigned)((ga >> 32) & 0x01FFFFFFu) | (2u << 30); // addr[56:32] | type=2
  u32x8 g1;
  g1[0] = (2u << 16);                                    // data_size=4B; mask=0; no pad
  g1[1] = (dim0 & 0xFFFFu) << 16;                        // tensor_dim0[15:0] @ [63:48]
  g1[2] = ((dim0 >> 16) & 0xFFFFu) | ((rows_total & 0xFFFFu) << 16); // dim0 hi | dim1 lo
  g1[3] = ((rows_total >> 16) & 0xFFFFu) | ((tile0 & 0xFFFFu) << 16); // dim1 hi | tile_dim0
  g1[4] = (tile1 & 0xFFFFu);                             // tile_dim1 ; tile_dim2=0
  g1[5] = dim0;                                          // tensor_dim0_stride[31:0]
  g1[6] = 0u;                                            // stride0 hi | dim1_stride lo
  g1[7] = 0u;                                            // dim1_stride hi
  u32x4 gz = {0u, 0u, 0u, 0u};                           // groups 2/3: zero (<=2D)
  asm volatile("tensor_load_to_lds %0, %1, %2, %3"
               :: "s"(g0), "s"(g1), "s"(gz), "s"(gz)
               : "memory");
}

// ---------------- kernels ----------------

__global__ void zero_kernel(f4* __restrict__ p, int n4) {
  int i = blockIdx.x * blockDim.x + threadIdx.x;
  if (i < n4) p[i] = (f4){0.f, 0.f, 0.f, 0.f};
}

// h = x @ lin1_w   (N x 128)(128 x 128), no bias
__global__ void __launch_bounds__(256)
node_gemm_kernel(const float* __restrict__ x, const float* __restrict__ w,
                 float* __restrict__ h, int nrows) {
  __shared__ _Float16 sX[16 * FDIM];     // 4 KB
  const int tid  = threadIdx.x;
  const int wave = tid >> 5;
  const int lane = tid & 31;
  const int n0   = wave * 16;
  const int g    = lane >> 4, n = lane & 15;
  const int CH   = FDIM / 4;             // float4 chunks per row

  v16h bw[4];                            // register-resident B fragments
#pragma unroll
  for (int s = 0; s < 4; ++s) bw[s] = load_b_frag_g(w, FDIM, 32 * s, n0, lane);

  const int ntiles = (nrows + 15) >> 4;
  for (int t = blockIdx.x; t < ntiles; t += gridDim.x) {
    const int row0 = t * 16;
    __syncthreads();
    for (int i = tid; i < 16 * CH; i += 256) {       // vectorized f32->f16 staging
      const int row = i / CH, c4 = (i % CH) * 4;
      const int r = min(row0 + row, nrows - 1);
      f4 v = *(const f4*)(x + (size_t)r * FDIM + c4);
      *(h4*)&sX[row * FDIM + c4] = cvt4(v);
    }
    __syncthreads();
    v8f acc = {};
#pragma unroll
    for (int s = 0; s < 4; ++s)
      acc = wmma_f16(load_a_frag_h(sX, FDIM, 32 * s, lane), bw[s], acc);

    if (row0 + 16 <= nrows) {                        // full tile: unguarded stores
#pragma unroll
      for (int v = 0; v < 8; ++v)
        h[(row0 + v + 8 * g) * FDIM + n0 + n] = acc[v];
    } else {
#pragma unroll
      for (int v = 0; v < 8; ++v) {
        const int m = v + 8 * g;
        if (row0 + m < nrows) h[(row0 + m) * FDIM + n0 + n] = acc[v];
      }
    }
  }
}

// Fused CFConv edge kernel; edge_attr tiles arrive via TDM (double-buffered).
__global__ void __launch_bounds__(256)
edge_kernel(const float* __restrict__ edge_attr, const float* __restrict__ edge_weight,
            const int* __restrict__ edge_index, int E_,
            const float* __restrict__ f1_w, const float* __restrict__ f1_b,
            const float* __restrict__ f2_w, const float* __restrict__ f2_b,
            const float* __restrict__ h, float* __restrict__ agg) {
  __shared__ float    sA[2][16 * RBFD];  // double-buffered edge_attr tiles, 8 KB
  __shared__ _Float16 sT[16 * FDIM];     // tanh intermediate, 4 KB
  __shared__ float    sC[16];
  __shared__ int      sSrc[16], sDst[16];

  const int tid  = threadIdx.x;
  const int wave = tid >> 5;
  const int lane = tid & 31;
  const int n0   = wave * 16;
  const int g    = lane >> 4, n = lane & 15;

  // register-resident weight fragments + per-lane biases
  v16h bf1[2], bf2[4];
#pragma unroll
  for (int s = 0; s < 2; ++s) bf1[s] = load_b_frag_g(f1_w, FDIM, 32 * s, n0, lane);
#pragma unroll
  for (int s = 0; s < 4; ++s) bf2[s] = load_b_frag_g(f2_w, FDIM, 32 * s, n0, lane);
  const float b1 = f1_b[n0 + n];
  const float b2 = f2_b[n0 + n];

  const unsigned ldsA0 = (unsigned)(size_t)(void*)&sA[0][0];
  const unsigned ldsA1 = (unsigned)(size_t)(void*)&sA[1][0];
  const int ntiles = E_ >> 4;

  // prologue: DMA first tile into buffer 0
  if (wave == 0 && blockIdx.x < ntiles)
    tdm_load_2d(ldsA0, edge_attr + (size_t)blockIdx.x * 16 * RBFD, RBFD, E_, RBFD, 16);

  int ibuf = 0;
  for (int t = blockIdx.x; t < ntiles; t += gridDim.x) {
    const int e0 = t * 16;
    const int tn = t + gridDim.x;
    __syncthreads();   // everyone is done with buf[ibuf^1] from the previous tile
    if (wave == 0) {
      if (tn < ntiles) {   // prefetch next tile into the other buffer
        tdm_load_2d(ibuf ? ldsA0 : ldsA1, edge_attr + (size_t)tn * 16 * RBFD,
                    RBFD, E_, RBFD, 16);
        __builtin_amdgcn_s_wait_tensorcnt((short)1);  // oldest (current tile) done
      } else {
        __builtin_amdgcn_s_wait_tensorcnt((short)0);
      }
    }
    if (tid < 16) {
      const float r = edge_weight[e0 + tid];
      sC[tid] = 0.5f * (__cosf(3.14159265358979f * r / CUT) + 1.0f) * (r < CUT ? 1.0f : 0.0f);
      sSrc[tid] = edge_index[e0 + tid];        // row 0 = src
      sDst[tid] = edge_index[E_ + e0 + tid];   // row 1 = dst
    }
    __syncthreads();   // current tile in LDS + per-tile scalars visible

    // GEMM1: (16 x 64) @ f1_w[:, n0:n0+16]
    v8f acc = {};
#pragma unroll
    for (int s = 0; s < 2; ++s)
      acc = wmma_f16(load_a_frag_f(sA[ibuf], RBFD, 32 * s, lane), bf1[s], acc);
#pragma unroll
    for (int v = 0; v < 8; ++v) {
      const int m = v + 8 * g;
      sT[m * FDIM + n0 + n] = (_Float16)tanhf(acc[v] + b1);
    }
    __syncthreads();

    // GEMM2: (16 x 128) @ f2_w[:, n0:n0+16]
    v8f acc2 = {};
#pragma unroll
    for (int s = 0; s < 4; ++s)
      acc2 = wmma_f16(load_a_frag_h(sT, FDIM, 32 * s, lane), bf2[s], acc2);

    // W tile done: gather h[src] (L2-resident), multiply, scatter-add to agg[dst]
#pragma unroll
    for (int v = 0; v < 8; ++v) {
      const int   m   = v + 8 * g;
      const float Wv  = (acc2[v] + b2) * sC[m];
      const int   src = sSrc[m], dst = sDst[m];
      const float msg = h[src * FDIM + n0 + n] * Wv;
      atomicAdd(&agg[dst * FDIM + n0 + n], msg);
    }
    ibuf ^= 1;
  }
}

// Faithful SeqConv scatter: agg[center] += seq_weight[diff+1] * h[atom_types[center]]
__global__ void __launch_bounds__(128)
seq_kernel(const int* __restrict__ seq_neighs, int M_,
           const int* __restrict__ atom_types,
           const float* __restrict__ seq_weight,
           const float* __restrict__ h, float* __restrict__ agg) {
  const int j = blockIdx.x;
  const int c = threadIdx.x;
  if (j >= M_) return;
  const int center = seq_neighs[j];
  const int nb     = seq_neighs[M_ + j];
  const int wi     = nb - center + 1;          // {0, 2}
  const int ntype  = atom_types[center];
  const float msg  = seq_weight[wi * FDIM + c] * h[ntype * FDIM + c];
  atomicAdd(&agg[center * FDIM + c], msg);
}

// Tail: t = agg + sw1 * h[atom_types];  out = tanh(t @ lin2_w + b2) @ blk_w + b3
__global__ void __launch_bounds__(256)
tail_kernel(const float* __restrict__ agg, const float* __restrict__ h,
            const int* __restrict__ atom_types, const float* __restrict__ seq_weight,
            const float* __restrict__ lin2_w, const float* __restrict__ lin2_b,
            const float* __restrict__ blk_w,  const float* __restrict__ blk_b,
            float* __restrict__ out, int nrows) {
  __shared__ _Float16 sIn[16 * FDIM];    // 4 KB
  __shared__ _Float16 sMid[16 * FDIM];   // 4 KB
  __shared__ float    sSW1[FDIM];

  const int tid  = threadIdx.x;
  const int wave = tid >> 5;
  const int lane = tid & 31;
  const int n0   = wave * 16;
  const int g    = lane >> 4, n = lane & 15;
  const int CH   = FDIM / 4;

  v16h bw2[4], bw3[4];
#pragma unroll
  for (int s = 0; s < 4; ++s) {
    bw2[s] = load_b_frag_g(lin2_w, FDIM, 32 * s, n0, lane);
    bw3[s] = load_b_frag_g(blk_w,  FDIM, 32 * s, n0, lane);
  }
  const float b2 = lin2_b[n0 + n];
  const float b3 = blk_b[n0 + n];
  if (tid < FDIM) sSW1[tid] = seq_weight[FDIM + tid];

  const int ntiles = (nrows + 15) >> 4;
  for (int t = blockIdx.x; t < ntiles; t += gridDim.x) {
    const int row0 = t * 16;
    __syncthreads();
    for (int i = tid; i < 16 * CH; i += 256) {       // vectorized input staging
      const int row = i / CH, c4 = (i % CH) * 4;
      const int node = min(row0 + row, nrows - 1);
      const int ty   = atom_types[node];
      f4 a  = *(const f4*)(agg + (size_t)node * FDIM + c4);
      f4 hh = *(const f4*)(h   + (size_t)ty   * FDIM + c4);
      f4 sw = *(const f4*)&sSW1[c4];
      f4 v; v[0] = a[0] + sw[0]*hh[0]; v[1] = a[1] + sw[1]*hh[1];
            v[2] = a[2] + sw[2]*hh[2]; v[3] = a[3] + sw[3]*hh[3];
      *(h4*)&sIn[row * FDIM + c4] = cvt4(v);
    }
    __syncthreads();

    v8f acc = {};
#pragma unroll
    for (int s = 0; s < 4; ++s)
      acc = wmma_f16(load_a_frag_h(sIn, FDIM, 32 * s, lane), bw2[s], acc);
#pragma unroll
    for (int v = 0; v < 8; ++v) {
      const int m = v + 8 * g;
      sMid[m * FDIM + n0 + n] = (_Float16)tanhf(acc[v] + b2);
    }
    __syncthreads();

    v8f acc2 = {};
#pragma unroll
    for (int s = 0; s < 4; ++s)
      acc2 = wmma_f16(load_a_frag_h(sMid, FDIM, 32 * s, lane), bw3[s], acc2);

    if (row0 + 16 <= nrows) {                        // full tile: unguarded stores
#pragma unroll
      for (int v = 0; v < 8; ++v)
        out[(row0 + v + 8 * g) * FDIM + n0 + n] = acc2[v] + b3;
    } else {
#pragma unroll
      for (int v = 0; v < 8; ++v) {
        const int m = v + 8 * g;
        if (row0 + m < nrows) out[(row0 + m) * FDIM + n0 + n] = acc2[v] + b3;
      }
    }
  }
}

// ---------------- launcher ----------------

extern "C" void kernel_launch(void* const* d_in, const int* in_sizes, int n_in,
                              void* d_out, int out_size, void* d_ws, size_t ws_size,
                              hipStream_t stream) {
  const float* x           = (const float*)d_in[0];
  const int*   edge_index  = (const int*)d_in[1];
  const float* edge_weight = (const float*)d_in[2];
  const float* edge_attr   = (const float*)d_in[3];
  const int*   atom_types  = (const int*)d_in[4];
  const int*   seq_neighs  = (const int*)d_in[5];
  const float* lin1_w      = (const float*)d_in[6];
  const float* f1_w        = (const float*)d_in[7];
  const float* f1_b        = (const float*)d_in[8];
  const float* f2_w        = (const float*)d_in[9];
  const float* f2_b        = (const float*)d_in[10];
  const float* seq_w       = (const float*)d_in[11];
  const float* lin2_w      = (const float*)d_in[12];
  const float* lin2_b      = (const float*)d_in[13];
  const float* blk_w       = (const float*)d_in[14];
  const float* blk_b       = (const float*)d_in[15];

  const int N_ = in_sizes[0] / FDIM;   // 20000
  const int E_ = in_sizes[2];          // 640000
  const int M_ = in_sizes[5] / 2;      // 2*(N-2)

  float* h   = (float*)d_ws;                    // N x 128
  float* agg = h + (size_t)N_ * FDIM;           // N x 128

  const int n4 = (N_ * FDIM) / 4;
  zero_kernel<<<(n4 + 255) / 256, 256, 0, stream>>>((f4*)agg, n4);
  node_gemm_kernel<<<256, 256, 0, stream>>>(x, lin1_w, h, N_);
  edge_kernel<<<1024, 256, 0, stream>>>(edge_attr, edge_weight, edge_index, E_,
                                        f1_w, f1_b, f2_w, f2_b, h, agg);
  seq_kernel<<<M_, 128, 0, stream>>>(seq_neighs, M_, atom_types, seq_w, h, agg);
  tail_kernel<<<256, 256, 0, stream>>>(agg, h, atom_types, seq_w,
                                       lin2_w, lin2_b, blk_w, blk_b,
                                       (float*)d_out, N_);
}
